// MyRNN_47656957116500
// MI455X (gfx1250) — matrix operand order repaired and checked by reading, not compiled, not run on hardware
//
#include <hip/hip_runtime.h>
#include <cmath>

typedef _Float16 h16;
typedef __attribute__((ext_vector_type(16))) _Float16 v16h;
typedef __attribute__((ext_vector_type(8)))  _Float16 v8h;
typedef __attribute__((ext_vector_type(8)))  float    v8f;

// Problem sizes (match reference)
#define SB   256   // batch
#define ST   512   // time steps
#define SDIN 128
#define SDST 128
#define SDH  512

#define BM   16    // batch rows per workgroup
#define NT1  32    // GEMM1 N tiles (512/16)
#define KC1  8     // GEMM1 K chunks (256/32)
#define KC2  16    // GEMM2 K chunks (512/32)

#define W1P_HALFS (KC1*NT1*512)   // 131072 halves = 256 KB
#define W2P_HALFS (KC2*8*512)     //  65536 halves = 128 KB

// LDS layout (dynamic): packed W1 | xa f16 [16][256] | h f16 [16][512] | a f32 [16][128]
#define LDS_W1   0
#define LDS_XA   (262144)
#define LDS_H    (262144 + 8192)
#define LDS_AF   (262144 + 8192 + 16384)
#define LDS_TOTAL (262144 + 8192 + 16384 + 8192)   // 294912 B < 320 KB

union A32 { v16h v; v8h h[2]; };

// -------- one-time weight packing: f32 row-major -> f16 WMMA B-operand blocks --------
// Block(kc, nt) covers K=[kc*32, kc*32+32), N=[nt*16, nt*16+16).
// Within a block, lane l's 16 halves are contiguous: half j of lane l is
// element (K = kc*32 + (l/16)*16 + j, N = nt*16 + l%16)  [same k order as the A loads].
__global__ void pack_weights(const float* __restrict__ W1,
                             const float* __restrict__ W2,
                             h16* __restrict__ w1p,
                             h16* __restrict__ w2p) {
  int idx = blockIdx.x * blockDim.x + threadIdx.x;
  if (idx < W1P_HALFS) {
    int j    = idx & 15;
    int lane = (idx >> 4) & 31;
    int blk  = idx >> 9;          // kc*NT1 + nt
    int nt   = blk & (NT1 - 1);
    int kc   = blk >> 5;
    int K = kc * 32 + (lane >> 4) * 16 + j;
    int N = nt * 16 + (lane & 15);
    w1p[idx] = (h16)W1[K * SDH + N];
  } else if (idx < W1P_HALFS + W2P_HALFS) {
    int i2   = idx - W1P_HALFS;
    int j    = i2 & 15;
    int lane = (i2 >> 4) & 31;
    int blk  = i2 >> 9;           // kc*8 + nt
    int nt   = blk & 7;
    int kc   = blk >> 3;
    int K = kc * 32 + (lane >> 4) * 16 + j;
    int N = nt * 16 + (lane & 15);
    w2p[i2] = (h16)W2[K * SDST + N];
  }
}

// -------- main recurrent kernel: one WG owns 16 batch rows for all 512 steps --------
__global__ __launch_bounds__(256)
void rnn_wmma(const float* __restrict__ x,    // [B,T,128]
              const float* __restrict__ a0,   // [B,128]
              const float* __restrict__ b1,   // [512]
              const float* __restrict__ b2,   // [128]
              const h16* __restrict__ w1p,
              const h16* __restrict__ w2p,
              float* __restrict__ out) {      // [B,128]
  extern __shared__ char smem[];
  h16*   w1l = (h16*)(smem + LDS_W1);
  h16*   xa  = (h16*)(smem + LDS_XA);   // [16][256]: cols 0..127 = x_t, 128..255 = a (f16)
  h16*   hL  = (h16*)(smem + LDS_H);    // [16][512]
  float* aF  = (float*)(smem + LDS_AF); // [16][128] f32 state

  const int tid  = threadIdx.x;
  const int wave = tid >> 5;
  const int lane = tid & 31;
  const int g    = lane >> 4;   // lane group
  const int ln   = lane & 15;
  const int m0   = blockIdx.x * BM;

  // W1 packed blocks: global -> LDS (one time), 16B vectors
  {
    const uint4* src = (const uint4*)w1p;
    uint4*       dst = (uint4*)w1l;
    for (int i = tid; i < (W1P_HALFS * 2) / 16; i += 256) dst[i] = src[i];
  }

  // W2 slice (my 16 output cols) -> 128 VGPRs, stationary for all steps
  v16h w2r[KC2];
#pragma unroll
  for (int kc = 0; kc < KC2; ++kc)
    w2r[kc] = *(const v16h*)(w2p + (kc * 8 + wave) * 512 + lane * 16);

  float b1r[4];
#pragma unroll
  for (int p = 0; p < 4; ++p) b1r[p] = b1[(wave * 4 + p) * 16 + ln];
  const float b2r = b2[wave * 16 + ln];

  // init f32 state + f16 copy in xa; load x_0
  for (int i = tid; i < BM * SDST; i += 256) {
    int r = i >> 7, c = i & 127;
    float v = a0[(m0 + r) * SDST + c];
    aF[i] = v;
    xa[r * 256 + 128 + c] = (h16)v;
  }
  for (int i = tid; i < BM * SDIN; i += 256) {
    int r = i >> 7, c = i & 127;
    xa[r * 256 + c] = (h16)x[((size_t)(m0 + r) * ST + 0) * SDIN + c];
  }
  __syncthreads();

  for (int t = 0; t < ST; ++t) {
    // overlap: prefetch next x tile into cache while WMMAs run
    if (t + 1 < ST)
      __builtin_prefetch(&x[((size_t)(m0 + (tid >> 4)) * ST + (t + 1)) * SDIN + (tid & 15) * 8], 0, 3);

    // ---- GEMM1: [16,256](xa) x [256,512](W1, LDS) -> per-wave [16,64] ----
    v8f acc0 = {0,0,0,0,0,0,0,0};
    v8f acc1 = acc0, acc2 = acc0, acc3 = acc0;
#pragma unroll
    for (int kc = 0; kc < KC1; ++kc) {
      A32 au;
      const int kb = kc * 32 + g * 16;
      au.h[0] = *(const v8h*)(xa + ln * 256 + kb);        // ds_load_b128
      au.h[1] = *(const v8h*)(xa + ln * 256 + kb + 8);    // ds_load_b128
      const h16* wb = w1l + (kc * NT1 + wave * 4) * 512 + lane * 16;
      v16h B0 = *(const v16h*)(wb);
      v16h B1 = *(const v16h*)(wb + 512);
      v16h B2 = *(const v16h*)(wb + 1024);
      v16h B3 = *(const v16h*)(wb + 1536);
      acc0 = __builtin_amdgcn_wmma_f32_16x16x32_f16(false, au.v, false, B0, (short)0, acc0, false, false);
      acc1 = __builtin_amdgcn_wmma_f32_16x16x32_f16(false, au.v, false, B1, (short)0, acc1, false, false);
      acc2 = __builtin_amdgcn_wmma_f32_16x16x32_f16(false, au.v, false, B2, (short)0, acc2, false, false);
      acc3 = __builtin_amdgcn_wmma_f32_16x16x32_f16(false, au.v, false, B3, (short)0, acc3, false, false);
    }
    // tanh + bias -> h (f16) in LDS.  C layout: lane -> (M = g*8+r, N = ln)
#pragma unroll
    for (int r = 0; r < 8; ++r) {
      const int mrow = g * 8 + r;
      hL[mrow * 512 + wave * 64 +  0 + ln] = (h16)tanhf(acc0[r] + b1r[0]);
      hL[mrow * 512 + wave * 64 + 16 + ln] = (h16)tanhf(acc1[r] + b1r[1]);
      hL[mrow * 512 + wave * 64 + 32 + ln] = (h16)tanhf(acc2[r] + b1r[2]);
      hL[mrow * 512 + wave * 64 + 48 + ln] = (h16)tanhf(acc3[r] + b1r[3]);
    }
    __syncthreads();

    // ---- GEMM2: [16,512](h) x [512,128](W2 in regs) -> per-wave [16,16] ----
    v8f c2 = {0,0,0,0,0,0,0,0};
#pragma unroll
    for (int kc = 0; kc < KC2; ++kc) {
      A32 au;
      const int kb = kc * 32 + g * 16;
      au.h[0] = *(const v8h*)(hL + ln * 512 + kb);
      au.h[1] = *(const v8h*)(hL + ln * 512 + kb + 8);
      c2 = __builtin_amdgcn_wmma_f32_16x16x32_f16(false, au.v, false, w2r[kc], (short)0, c2, false, false);
    }
    // a += h@W2 + b2 : f32 master copy in LDS, f16 copy for next step's A operand
#pragma unroll
    for (int r = 0; r < 8; ++r) {
      const int mrow = g * 8 + r;
      const int col  = wave * 16 + ln;
      float an = aF[mrow * 128 + col] + c2[r] + b2r;
      aF[mrow * 128 + col] = an;
      xa[mrow * 256 + 128 + col] = (h16)an;
    }
    // stage x_{t+1} (f32 global -> f16 LDS); safe: GEMM1 reads finished at barrier above
    if (t + 1 < ST) {
      for (int i = tid; i < BM * SDIN; i += 256) {
        int r = i >> 7, c = i & 127;
        xa[r * 256 + c] = (h16)x[((size_t)(m0 + r) * ST + (t + 1)) * SDIN + c];
      }
    }
    __syncthreads();
  }

  // write final f32 state
  for (int i = tid; i < BM * SDST; i += 256)
    out[(m0 + (i >> 7)) * SDST + (i & 127)] = aF[i];
}

extern "C" void kernel_launch(void* const* d_in, const int* in_sizes, int n_in,
                              void* d_out, int out_size, void* d_ws, size_t ws_size,
                              hipStream_t stream) {
  (void)in_sizes; (void)n_in; (void)out_size; (void)ws_size;
  const float* x  = (const float*)d_in[0];
  const float* a0 = (const float*)d_in[1];
  const float* W1 = (const float*)d_in[2];
  const float* b1 = (const float*)d_in[3];
  const float* W2 = (const float*)d_in[4];
  const float* b2 = (const float*)d_in[5];
  float* out = (float*)d_out;

  h16* w1p = (h16*)d_ws;                               // 256 KB
  h16* w2p = (h16*)((char*)d_ws + W1P_HALFS * 2);      // 128 KB

  const int packThreads = W1P_HALFS + W2P_HALFS;
  pack_weights<<<(packThreads + 255) / 256, 256, 0, stream>>>(W1, W2, w1p, w2p);
  rnn_wmma<<<SB / BM, 256, LDS_TOTAL, stream>>>(x, a0, b1, b2, w1p, w2p, out);
}
